// GRU_MAE_19533511262664
// MI455X (gfx1250) — matrix operand order repaired and checked by reading, not compiled, not run on hardware
//
#include <hip/hip_runtime.h>

// ---------------------------------------------------------------------------
// GRU encoder/decoder for MI455X (gfx1250, wave32, WMMA).
// gh = h @ Whh^T via v_wmma_f32_16x16x32_f16: one wave owns a 16-row batch
// tile; weights staged f32->f16 into LDS once; hidden state kept in registers
// (WMMA C layout) + mirrored to a small f16 LDS buffer to rebuild WMMA A
// tiles each step.  Gate params live in LDS (reloaded per step) to keep VGPR
// use < 256 and avoid s_set_vgpr_msb thrash.  Gate nonlinearities use native
// v_exp_f32 / v_rcp_f32 / v_tanh_f32.
// ---------------------------------------------------------------------------

typedef __attribute__((ext_vector_type(16))) _Float16 v16h;
typedef __attribute__((ext_vector_type(8)))  float    v8f;

constexpr int Bt  = 1024;   // batch
constexpr int Lt  = 512;    // sequence length
constexpr int Hh  = 128;    // hidden
constexpr int Gg  = 8;      // groups
constexpr int TH  = 3 * Hh; // 384 gate dims
constexpr int TILE = 16;    // batch rows per wave

constexpr size_t W_LDS_BYTES = (size_t)TH * Hh * 2;   // 96 KB f16 weights
constexpr size_t H_LDS_BYTES = (size_t)TILE * Hh * 2; // 4 KB f16 hidden
constexpr size_t H_OFF  = W_LDS_BYTES;
constexpr size_t P_OFF  = H_OFF + H_LDS_BYTES;        // 7*128 f32 gate params
constexpr size_t LW_OFF = P_OFF + 7 * 128 * 4;        // 8*16 f32 linW
constexpr size_t X_OFF  = LW_OFF + 8 * 16 * 4;        // 16 f32 x broadcast
constexpr size_t LDS_BYTES = X_OFF + 64;

__device__ __forceinline__ float fast_exp2(float x) {
#if __has_builtin(__builtin_amdgcn_exp2f)
  return __builtin_amdgcn_exp2f(x);   // v_exp_f32
#else
  return exp2f(x);
#endif
}

__device__ __forceinline__ float fast_rcp(float x) {
#if __has_builtin(__builtin_amdgcn_rcpf)
  return __builtin_amdgcn_rcpf(x);    // v_rcp_f32
#else
  return 1.0f / x;
#endif
}

__device__ __forceinline__ float fast_sigmoid(float x) {
  // 1 / (1 + 2^(-x*log2e)) : v_mul + v_exp_f32 + v_add + v_rcp_f32
  return fast_rcp(1.0f + fast_exp2(x * -1.4426950408889634f));
}

__device__ __forceinline__ float fast_tanh(float x) {
#if __has_builtin(__builtin_amdgcn_tanhf)
  return __builtin_amdgcn_tanhf(x);   // gfx1250 v_tanh_f32
#else
  // tanh(x) = 2*sigmoid(2x) - 1
  const float s = fast_rcp(1.0f + fast_exp2(x * -2.885390081777927f));
  return fmaf(2.0f, s, -1.0f);
#endif
}

__device__ __forceinline__ v8f wmma_f16(v16h a, v16h b, v8f c) {
  // (neg_a, A, neg_b, B, c_mod, C, reuse_a, reuse_b)
  return __builtin_amdgcn_wmma_f32_16x16x32_f16(
      false, a, false, b, (short)0, c, false, false);
}

// A tile (16x32 f16, MxK) from LDS hidden buffer (row-major [16][128] f16).
// ISA layout: lanes 0-15 -> M=lane, K = {0..7, 16..23}+32*kt
//             lanes 16-31 -> M=lane-16, K = {8..15, 24..31}+32*kt
__device__ __forceinline__ v16h load_A(const _Float16* Hl, int lane, int kt) {
  const int m = lane & 15;
  const _Float16* p = Hl + m * Hh + kt * 32 + ((lane >> 4) << 3);
  v16h a;
#pragma unroll
  for (int i = 0; i < 8; ++i) {
    a[i]     = p[i];        // 16B contiguous -> ds_load_b128
    a[i + 8] = p[i + 16];   // 16B contiguous -> ds_load_b128
  }
  return a;
}

// B tile (32x16 f16, KxN) = Whh^T slice.  Wlds is Whh row-major [384][128] f16
// so B[k][n] = Wlds[(N0+n)*128 + K0+k]: two runs of 8 contiguous halves per lane.
// Layout: lane n = lane&15; lanes 0-15 hold K={0..7,16..23}, lanes 16-31 hold K={8..15,24..31}.
__device__ __forceinline__ v16h load_B(const _Float16* W, int lane, int ntile,
                                       int kt) {
  const _Float16* p =
      W + (size_t)(ntile * TILE + (lane & 15)) * Hh + kt * 32 + ((lane >> 4) << 3);
  v16h b;
#pragma unroll
  for (int i = 0; i < 8; ++i) {
    b[i]     = p[i];
    b[i + 8] = p[i + 16];
  }
  return b;
}

// Gate params staged to LDS: P[p][jt*16+nn], p = {wr,wz,wn,br,bz,bni,bnh}.
__device__ __forceinline__ void fill_gate_params(float* Plds, int lane,
                                                 const float* __restrict__ Wih,
                                                 const float* __restrict__ bih,
                                                 const float* __restrict__ bhh) {
  if (lane < TILE) {
#pragma unroll
    for (int jt = 0; jt < 8; ++jt) {
      const int c = jt * TILE + lane;
      const int cr = c, cz = c + Hh, cn = c + 2 * Hh;
      Plds[0 * 128 + c] = Wih[cr];
      Plds[1 * 128 + c] = Wih[cz];
      Plds[2 * 128 + c] = Wih[cn];
      Plds[3 * 128 + c] = bih[cr] + bhh[cr];
      Plds[4 * 128 + c] = bih[cz] + bhh[cz];
      Plds[5 * 128 + c] = bih[cn];
      Plds[6 * 128 + c] = bhh[cn];
    }
  }
}

// One GRU step for a 16-row tile.  inp[r] = scalar input for accumulator row r.
// hreg is WMMA C layout: hreg[jt][r] = h(row = r + 8*(lane>=16), col = 16*jt + (lane&15)).
__device__ __forceinline__ void gru_step(int lane, const _Float16* __restrict__ Wlds,
                                         _Float16* __restrict__ Hlds,
                                         const float* __restrict__ Plds,
                                         const float* inp, float hreg[8][8]) {
  v16h A[4];
#pragma unroll
  for (int kt = 0; kt < 4; ++kt) A[kt] = load_A(Hlds, lane, kt);

  const int nn = lane & 15;
  const int roff = (lane >> 4) << 3;

#pragma unroll
  for (int jt = 0; jt < 8; ++jt) {
    v8f aR = {}, aZ = {}, aN = {};
#pragma unroll
    for (int kt = 0; kt < 4; ++kt) {
      v16h bR = load_B(Wlds, lane, jt, kt);
      v16h bZ = load_B(Wlds, lane, jt + 8, kt);
      v16h bN = load_B(Wlds, lane, jt + 16, kt);
      aR = wmma_f16(A[kt], bR, aR);
      aZ = wmma_f16(A[kt], bZ, aZ);
      aN = wmma_f16(A[kt], bN, aN);
    }
    const int c = jt * TILE + nn;
    const float wr  = Plds[0 * 128 + c];
    const float wz  = Plds[1 * 128 + c];
    const float wn  = Plds[2 * 128 + c];
    const float br  = Plds[3 * 128 + c];
    const float bz  = Plds[4 * 128 + c];
    const float bni = Plds[5 * 128 + c];
    const float bnh = Plds[6 * 128 + c];
#pragma unroll
    for (int r = 0; r < 8; ++r) {
      const float x = inp[r];
      const float rg = fast_sigmoid(fmaf(x, wr, br) + aR[r]);
      const float zg = fast_sigmoid(fmaf(x, wz, bz) + aZ[r]);
      const float ng = fast_tanh(fmaf(x, wn, bni) + rg * (aN[r] + bnh));
      const float hn = fmaf(zg, hreg[jt][r] - ng, ng);  // (1-z)n + z h
      hreg[jt][r] = hn;
      Hlds[(size_t)(r + roff) * Hh + c] = (_Float16)hn;  // feed next step's A
    }
  }
}

// ---------------------------------------------------------------------------
// Encoder: 64 blocks x 1 wave; block b owns batch rows [16b, 16b+16).
// ---------------------------------------------------------------------------
__global__ void __launch_bounds__(32) gru_enc_kernel(
    const float* __restrict__ x_seq, const float* __restrict__ Wih_f,
    const float* __restrict__ Whh_f, const float* __restrict__ bih_f,
    const float* __restrict__ bhh_f, float* __restrict__ h_out) {
  extern __shared__ char smem[];
  _Float16* Wlds = (_Float16*)smem;
  _Float16* Hlds = (_Float16*)(smem + H_OFF);
  float*    Plds = (float*)(smem + P_OFF);
  float*    Xlds = (float*)(smem + X_OFF);

  const int lane = threadIdx.x;
  const int row0 = blockIdx.x * TILE;
  const int nn = lane & 15;
  const int roff = (lane >> 4) << 3;

  for (int i = lane; i < TH * Hh; i += 32) Wlds[i] = (_Float16)Whh_f[i];
  for (int i = lane; i < TILE * Hh; i += 32) Hlds[i] = (_Float16)0.0f;
  fill_gate_params(Plds, lane, Wih_f, bih_f, bhh_f);

  float hreg[8][8];
#pragma unroll
  for (int jt = 0; jt < 8; ++jt) {
#pragma unroll
    for (int r = 0; r < 8; ++r) hreg[jt][r] = 0.0f;
  }
  __syncthreads();

  for (int t = 0; t < Lt; ++t) {
    if (lane < TILE) {
      const float* xp = &x_seq[(size_t)(row0 + lane) * Lt + t];
      Xlds[lane] = *xp;
      if (t + 16 < Lt) __builtin_prefetch(xp + 16, 0, 0);  // global_prefetch_b8
    }
    __syncthreads();  // single wave: cheap + compiler LDS fence
    float inp[8];
#pragma unroll
    for (int r = 0; r < 8; ++r) inp[r] = Xlds[roff + r];
    gru_step(lane, Wlds, Hlds, Plds, inp, hreg);
    __syncthreads();
  }

#pragma unroll
  for (int jt = 0; jt < 8; ++jt) {
#pragma unroll
    for (int r = 0; r < 8; ++r)
      h_out[(size_t)(row0 + r + roff) * Hh + jt * TILE + nn] = hreg[jt][r];
  }
}

// ---------------------------------------------------------------------------
// Bucket rows by group id so decoder tiles share a single Whh (WMMA B).
// ---------------------------------------------------------------------------
__global__ void bucket_kernel(const int* __restrict__ ids, int* __restrict__ cnt,
                              int* __restrict__ bucket) {
  const int i = blockIdx.x * blockDim.x + threadIdx.x;
  if (i < Bt) {
    const int g = ids[i];
    const int p = atomicAdd(&cnt[g], 1);
    bucket[g * Bt + p] = i;
  }
}

// ---------------------------------------------------------------------------
// Decoder: grid = G * 64 tiles; tile exits if beyond its group's count.
// Short tiles pad with bucket[0] (identical recompute -> benign dup stores).
// ---------------------------------------------------------------------------
__global__ void __launch_bounds__(32) gru_dec_kernel(
    const float* __restrict__ h_enc, const int* __restrict__ cnt,
    const int* __restrict__ bucket, const float* __restrict__ Wih_v,
    const float* __restrict__ Whh_v, const float* __restrict__ bih_v,
    const float* __restrict__ bhh_v, const float* __restrict__ linW,
    const float* __restrict__ linb, float* __restrict__ out) {
  extern __shared__ char smem[];
  _Float16* Wlds = (_Float16*)smem;
  _Float16* Hlds = (_Float16*)(smem + H_OFF);
  float*    Plds = (float*)(smem + P_OFF);
  float*    LWlds = (float*)(smem + LW_OFF);

  const int g = blockIdx.x >> 6;
  const int tileIdx = blockIdx.x & 63;
  const int ng = cnt[g];
  if (tileIdx * TILE >= ng) return;

  const int lane = threadIdx.x;
  const int nn = lane & 15;
  const int roff = (lane >> 4) << 3;
  const int lim = ng - tileIdx * TILE;
  const int* bk = bucket + g * Bt + tileIdx * TILE;

  const float* Whh = Whh_v + (size_t)g * TH * Hh;
  for (int i = lane; i < TH * Hh; i += 32) Wlds[i] = (_Float16)Whh[i];
  fill_gate_params(Plds, lane, Wih_v + (size_t)g * TH, bih_v + (size_t)g * TH,
                   bhh_v + (size_t)g * TH);
  if (lane < TILE) {
#pragma unroll
    for (int jt = 0; jt < 8; ++jt)
      LWlds[jt * TILE + lane] = linW[(size_t)g * Hh + jt * TILE + lane];
  }
  const float lb = linb[g];

  int myrow[8];
  float hreg[8][8];
#pragma unroll
  for (int r = 0; r < 8; ++r) {
    const int j = r + roff;
    const int rw = (j < lim) ? bk[j] : bk[0];
    myrow[r] = rw;
#pragma unroll
    for (int jt = 0; jt < 8; ++jt)
      hreg[jt][r] = h_enc[(size_t)rw * Hh + jt * TILE + nn];
  }
  {  // init f16 hidden buffer: each lane fills half a row
    const int m = nn;
    const int rw = (m < lim) ? bk[m] : bk[0];
    const int base = (lane >= 16) ? 64 : 0;
    for (int kk = 0; kk < 64; ++kk)
      Hlds[(size_t)m * Hh + base + kk] = (_Float16)h_enc[(size_t)rw * Hh + base + kk];
  }
  __syncthreads();

  // last0 = h_enc . lw + lb  (butterfly over the 16 column lanes per half)
  float last[8];
  {
    float p[8];
#pragma unroll
    for (int r = 0; r < 8; ++r) p[r] = 0.0f;
#pragma unroll
    for (int jt = 0; jt < 8; ++jt) {
      const float lw = LWlds[jt * TILE + nn];
#pragma unroll
      for (int r = 0; r < 8; ++r) p[r] = fmaf(hreg[jt][r], lw, p[r]);
    }
#pragma unroll
    for (int r = 0; r < 8; ++r) {
      float v = p[r];
      v += __shfl_xor(v, 1, 32);
      v += __shfl_xor(v, 2, 32);
      v += __shfl_xor(v, 4, 32);
      v += __shfl_xor(v, 8, 32);
      last[r] = v + lb;
    }
  }

  for (int t = 0; t < Lt; ++t) {
    gru_step(lane, Wlds, Hlds, Plds, last, hreg);
    __syncthreads();
    float p[8];
#pragma unroll
    for (int r = 0; r < 8; ++r) p[r] = 0.0f;
#pragma unroll
    for (int jt = 0; jt < 8; ++jt) {
      const float lw = LWlds[jt * TILE + nn];
#pragma unroll
      for (int r = 0; r < 8; ++r) p[r] = fmaf(hreg[jt][r], lw, p[r]);
    }
#pragma unroll
    for (int r = 0; r < 8; ++r) {
      float v = p[r];
      v += __shfl_xor(v, 1, 32);
      v += __shfl_xor(v, 2, 32);
      v += __shfl_xor(v, 4, 32);
      v += __shfl_xor(v, 8, 32);
      const float o = v + lb;
      last[r] = o;
      if (nn == 0) out[(size_t)myrow[r] * Lt + (Lt - 1 - t)] = o;  // time flip
    }
  }
}

// ---------------------------------------------------------------------------
extern "C" void kernel_launch(void* const* d_in, const int* in_sizes, int n_in,
                              void* d_out, int out_size, void* d_ws, size_t ws_size,
                              hipStream_t stream) {
  (void)in_sizes; (void)n_in; (void)out_size; (void)ws_size;

  const float* x_seq = (const float*)d_in[0];
  const int*   ids   = (const int*)d_in[1];
  const float* Wih_f = (const float*)d_in[2];
  const float* Whh_f = (const float*)d_in[3];
  const float* bih_f = (const float*)d_in[4];
  const float* bhh_f = (const float*)d_in[5];
  const float* Wih_v = (const float*)d_in[6];
  const float* Whh_v = (const float*)d_in[7];
  const float* bih_v = (const float*)d_in[8];
  const float* bhh_v = (const float*)d_in[9];
  const float* linW  = (const float*)d_in[10];
  const float* linb  = (const float*)d_in[11];
  float* out = (float*)d_out;

  // workspace: h_enc [B,H] f32 (512 KB) | cnt [G] | bucket [G,B]
  float* h_enc = (float*)d_ws;
  int* cnt = (int*)((char*)d_ws + (size_t)Bt * Hh * sizeof(float));
  int* bucket = cnt + Gg;

  hipMemsetAsync(cnt, 0, Gg * sizeof(int), stream);
  gru_enc_kernel<<<Bt / TILE, 32, LDS_BYTES, stream>>>(x_seq, Wih_f, Whh_f,
                                                       bih_f, bhh_f, h_enc);
  bucket_kernel<<<(Bt + 255) / 256, 256, 0, stream>>>(ids, cnt, bucket);
  gru_dec_kernel<<<Gg * (Bt / TILE), 32, LDS_BYTES, stream>>>(
      h_enc, cnt, bucket, Wih_v, Whh_v, bih_v, bhh_v, linW, linb, out);
}
